// CDP_78099685310666
// MI455X (gfx1250) — matrix operand order repaired and checked by reading, not compiled, not run on hardware
//
#include <hip/hip_runtime.h>

typedef __attribute__((ext_vector_type(16))) _Float16 v16h;
typedef __attribute__((ext_vector_type(8)))  _Float16 v8h;
typedef __attribute__((ext_vector_type(2)))  _Float16 v2h;
typedef __attribute__((ext_vector_type(2)))  __fp16   v2hh;   // cvt_pkrtz result type
typedef __attribute__((ext_vector_type(8)))  float    v8f;
typedef __attribute__((ext_vector_type(4)))  float    v4f;

#define DFEAT 256
#define WH_STRIDE 264          // halves per LDS row of W (256 + 8 pad -> conflict-free)
#define ROWS_PER_BLOCK 128     // 8 waves * 16 rows

static __device__ __forceinline__ v2h pk_f16(float a, float b) {
  v2hh r = __builtin_amdgcn_cvt_pkrtz(a, b);   // v_cvt_pk_rtz_f16_f32
  return __builtin_bit_cast(v2h, r);
}

__global__ __launch_bounds__(256) void fused_cosgate_kernel(
    const float* __restrict__ pred, const float* __restrict__ later,
    const float* __restrict__ W, const float* __restrict__ bias,
    float* __restrict__ out, int nRows)
{
  // f16 copy of W[:, :256] staged in LDS (132 KB; 2 workgroups fit in 320 KB WGP LDS)
  __shared__ _Float16 sW[DFEAT * WH_STRIDE];

  const int tid = threadIdx.x;

  // ---- stage W -> LDS as f16 (cols 0..255; col 256 handled separately) ----
  {
    const int rowT = tid >> 6;           // 0..3
    const int colT = (tid & 63) * 4;     // 0..252
    for (int r0 = 0; r0 < DFEAT; r0 += 4) {
      const int r = r0 + rowT;
      const float* wp = W + (size_t)r * (DFEAT + 1) + colT;  // row stride 257 (unaligned -> scalar loads)
      float w0 = wp[0], w1 = wp[1], w2 = wp[2], w3 = wp[3];
      v2h lo = pk_f16(w0, w1);
      v2h hi = pk_f16(w2, w3);
      *(v2h*)(&sW[r * WH_STRIDE + colT])     = lo;
      *(v2h*)(&sW[r * WH_STRIDE + colT + 2]) = hi;
    }
  }
  __syncthreads();

  const int lane = tid & 31;           // wave32
  const int wv   = tid >> 5;           // 0..7
  const int mloc = lane & 15;          // row-in-tile this lane owns
  const int hlf  = lane >> 4;          // 0 or 1 (which half of K / which M group)
  const int rowBase = blockIdx.x * ROWS_PER_BLOCK + wv * 16;
  if (rowBase + 16 > nRows) return;    // wave-uniform: EXEC stays all-ones for WMMA

  // ---- cosine distance per row: dis = 1 - <p,l> / (max(|p|,eps)*max(|l|,eps)) ----
  float spp = 0.f, sll = 0.f, spl = 0.f;
  {
    const float* pr = pred  + (size_t)(rowBase + mloc) * DFEAT + hlf * 128;
    const float* lr = later + (size_t)(rowBase + mloc) * DFEAT + hlf * 128;
    #pragma unroll
    for (int i = 0; i < 128; i += 4) {
      v4f p = *(const v4f*)(pr + i);
      v4f q = *(const v4f*)(lr + i);
      #pragma unroll
      for (int j = 0; j < 4; ++j) {
        spp = fmaf(p[j], p[j], spp);
        sll = fmaf(q[j], q[j], sll);
        spl = fmaf(p[j], q[j], spl);
      }
    }
  }
  spp += __shfl_xor(spp, 16, 32);
  sll += __shfl_xor(sll, 16, 32);
  spl += __shfl_xor(spl, 16, 32);
  const float eps = 1e-12f;
  const float denom = fmaxf(sqrtf(spp), eps) * fmaxf(sqrtf(sll), eps);
  const float dis = 1.0f - spl / denom;

  // dis for accumulator row M = v + 8*hlf lives in lane (v + 8*hlf) -> ds_bpermute
  float disv[8];
  #pragma unroll
  for (int v = 0; v < 8; ++v)
    disv[v] = __shfl(dis, (hlf << 3) + v, 32);

  // ---- A fragments: fea_later rows as f16 in WMMA 16x32 A layout, all K kept in VGPRs ----
  v16h afr[8];
  {
    const float* lrow = later + (size_t)(rowBase + mloc) * DFEAT;
    #pragma unroll
    for (int s = 0; s < 8; ++s) {
      const int c0 = 32 * s + (hlf << 3);
      v4f a0 = *(const v4f*)(lrow + c0);
      v4f a1 = *(const v4f*)(lrow + c0 + 4);
      v4f a2 = *(const v4f*)(lrow + c0 + 16);
      v4f a3 = *(const v4f*)(lrow + c0 + 20);
      v2h p0 = pk_f16(a0[0], a0[1]);
      v2h p1 = pk_f16(a0[2], a0[3]);
      v2h p2 = pk_f16(a1[0], a1[1]);
      v2h p3 = pk_f16(a1[2], a1[3]);
      v2h p4 = pk_f16(a2[0], a2[1]);
      v2h p5 = pk_f16(a2[2], a2[3]);
      v2h p6 = pk_f16(a3[0], a3[1]);
      v2h p7 = pk_f16(a3[2], a3[3]);
      v16h f;
      f[0]=p0[0];  f[1]=p0[1];  f[2]=p1[0];  f[3]=p1[1];
      f[4]=p2[0];  f[5]=p2[1];  f[6]=p3[0];  f[7]=p3[1];
      f[8]=p4[0];  f[9]=p4[1];  f[10]=p5[0]; f[11]=p5[1];
      f[12]=p6[0]; f[13]=p6[1]; f[14]=p7[0]; f[15]=p7[1];
      afr[s] = f;
    }
  }

  // ---- 16 N-tiles of 16 output features; K = 256 = 8 x WMMA_16x16x32 ----
  #pragma unroll 1
  for (int t = 0; t < 16; ++t) {
    const int n = t * 16 + mloc;                       // output feature for this lane
    const float bn = bias[n];
    const float wl = W[(size_t)n * (DFEAT + 1) + DFEAT]; // 257th column: the dis term

    v8f acc;
    #pragma unroll
    for (int v = 0; v < 8; ++v) acc[v] = fmaf(disv[v], wl, bn);

    const _Float16* bp = &sW[(size_t)n * WH_STRIDE + (hlf << 3)];
    #pragma unroll
    for (int s = 0; s < 8; ++s) {
      v8h b0 = *(const v8h*)(bp + 32 * s);
      v8h b1 = *(const v8h*)(bp + 32 * s + 16);
      v16h bfr = __builtin_shufflevector(b0, b1,
                   0,1,2,3,4,5,6,7,8,9,10,11,12,13,14,15);
      acc = __builtin_amdgcn_wmma_f32_16x16x32_f16(
                false, afr[s], false, bfr, (short)0, acc, false, false);
    }

    // epilogue: out = fea_later * (1 + sigmoid(acc)); fast rcp-based sigmoid
    #pragma unroll
    for (int v = 0; v < 8; ++v) {
      const size_t idx = (size_t)(rowBase + v + (hlf << 3)) * DFEAT + n;
      const float fl  = later[idx];
      const float wgt = __builtin_amdgcn_rcpf(1.0f + __expf(-acc[v]));
      out[idx] = fmaf(fl, wgt, fl);
    }
  }
}

extern "C" void kernel_launch(void* const* d_in, const int* in_sizes, int n_in,
                              void* d_out, int out_size, void* d_ws, size_t ws_size,
                              hipStream_t stream) {
  const float* pred  = (const float*)d_in[0];   // fea_pred  (L,B,256) f32
  const float* later = (const float*)d_in[1];   // fea_later (L,B,256) f32
  const float* W     = (const float*)d_in[2];   // (256,257) f32
  const float* bias  = (const float*)d_in[3];   // (256,)    f32
  float* out = (float*)d_out;

  const int nRows = in_sizes[0] / DFEAT;        // L*B = 262144
  const int blocks = (nRows + ROWS_PER_BLOCK - 1) / ROWS_PER_BLOCK;

  fused_cosgate_kernel<<<blocks, 256, 0, stream>>>(pred, later, W, bias, out, nRows);
}
